// MixHop_88450556494349
// MI455X (gfx1250) — compile-verified, matrix operand order checked by reading, not run on hardware
//
#include <hip/hip_runtime.h>
#include <cstdint>
#include <cstddef>

// ---------------------------------------------------------------------------
// Types for CDNA5 WMMA (wave32): D(16x16 f32) = A(16x32 bf16) x B(32x16 bf16) + C
// ---------------------------------------------------------------------------
typedef __bf16 bf16_t;
typedef __attribute__((ext_vector_type(16))) __bf16 v16bf;
typedef __attribute__((ext_vector_type(8)))  __bf16 v8bf;
typedef __attribute__((ext_vector_type(8)))  float  v8f;

__device__ __forceinline__ unsigned short f2bf(float f) {
  unsigned int u = __float_as_uint(f);
  u += 0x7fffu + ((u >> 16) & 1u);            // round-to-nearest-even
  return (unsigned short)(u >> 16);
}

// A/B fragment for 16x16x32 bf16 WMMA, loaded from row-major memory.
// Per ISA layout: lane l holds row (l&15); element i maps to
// K = (i%8) + (i/8)*16 + (l>>4)*8  -> two contiguous 16B loads per lane.
__device__ __forceinline__ v16bf load_frag(const bf16_t* p) {
  v8bf lo = *reinterpret_cast<const v8bf*>(p);        // K: base+0..7
  v8bf hi = *reinterpret_cast<const v8bf*>(p + 16);   // K: base+16..23
  return __builtin_shufflevector(lo, hi, 0, 1, 2, 3, 4, 5, 6, 7,
                                         8, 9, 10, 11, 12, 13, 14, 15);
}

// ---------------------------------------------------------------------------
// Utility kernels
// ---------------------------------------------------------------------------
__global__ void __launch_bounds__(256) fill_f32(float* __restrict__ p,
                                                long long n, float v) {
  long long i = (long long)blockIdx.x * blockDim.x + threadIdx.x;
  long long s = (long long)gridDim.x * blockDim.x;
  for (; i < n; i += s) p[i] = v;
}

__global__ void __launch_bounds__(256) cvt_f32_bf16(const float* __restrict__ in,
                                                    unsigned short* __restrict__ out,
                                                    long long n) {
  long long i = (long long)blockIdx.x * blockDim.x + threadIdx.x;
  long long s = (long long)gridDim.x * blockDim.x;
  for (; i < n; i += s) out[i] = f2bf(in[i]);
}

// ---------------------------------------------------------------------------
// bf16 WMMA GEMM:  C[M x 128] = alpha * (A[M x K](bf16) @ W[128 x K]^T + bias)
// Block: 256 threads = 8 waves; block tile 128(M) x 128(N).
// Wave (mgrp in 0..3, ngrp in 0..1): 2 M-tiles x 4 N-tiles = 8 v8f accums.
// ---------------------------------------------------------------------------
__global__ void __launch_bounds__(256) gemm_bf16_n128(
    const bf16_t* __restrict__ A, const bf16_t* __restrict__ W,
    const float* __restrict__ bias, float* __restrict__ C,
    int M, int K, int ldc, int coff, float alpha)
{
  const int lane = threadIdx.x & 31;
  const int wave = threadIdx.x >> 5;
  const int mgrp = wave & 3;            // 4 row-groups of 32
  const int ngrp = wave >> 2;           // 2 col-groups of 64
  const int rowBase = blockIdx.x * 128 + mgrp * 32;
  const int colBase = ngrp * 64;
  const int half = lane >> 4;
  const int ml   = lane & 15;

  v8f acc[2][4] = {};

  // Clamp A rows so every lane stays in-bounds (EXEC must be all-1s for WMMA).
  int r0 = rowBase + ml;      if (r0 > M - 1) r0 = M - 1;
  int r1 = rowBase + 16 + ml; if (r1 > M - 1) r1 = M - 1;
  const bf16_t* a0 = A + (size_t)r0 * K + half * 8;
  const bf16_t* a1 = A + (size_t)r1 * K + half * 8;
  const bf16_t* w0 = W + (size_t)(colBase +  0 + ml) * K + half * 8;
  const bf16_t* w1 = W + (size_t)(colBase + 16 + ml) * K + half * 8;
  const bf16_t* w2 = W + (size_t)(colBase + 32 + ml) * K + half * 8;
  const bf16_t* w3 = W + (size_t)(colBase + 48 + ml) * K + half * 8;

  for (int k = 0; k < K; k += 32) {
    v16bf af0 = load_frag(a0 + k);
    v16bf af1 = load_frag(a1 + k);
    v16bf bf0 = load_frag(w0 + k);
    v16bf bf1 = load_frag(w1 + k);
    v16bf bf2 = load_frag(w2 + k);
    v16bf bf3 = load_frag(w3 + k);
    acc[0][0] = __builtin_amdgcn_wmma_f32_16x16x32_bf16(false, af0, false, bf0, (short)0, acc[0][0], false, false);
    acc[0][1] = __builtin_amdgcn_wmma_f32_16x16x32_bf16(false, af0, false, bf1, (short)0, acc[0][1], false, false);
    acc[0][2] = __builtin_amdgcn_wmma_f32_16x16x32_bf16(false, af0, false, bf2, (short)0, acc[0][2], false, false);
    acc[0][3] = __builtin_amdgcn_wmma_f32_16x16x32_bf16(false, af0, false, bf3, (short)0, acc[0][3], false, false);
    acc[1][0] = __builtin_amdgcn_wmma_f32_16x16x32_bf16(false, af1, false, bf0, (short)0, acc[1][0], false, false);
    acc[1][1] = __builtin_amdgcn_wmma_f32_16x16x32_bf16(false, af1, false, bf1, (short)0, acc[1][1], false, false);
    acc[1][2] = __builtin_amdgcn_wmma_f32_16x16x32_bf16(false, af1, false, bf2, (short)0, acc[1][2], false, false);
    acc[1][3] = __builtin_amdgcn_wmma_f32_16x16x32_bf16(false, af1, false, bf3, (short)0, acc[1][3], false, false);
  }

  // Epilogue: C/D layout -> lane l holds col (l&15), VGPR v holds row v + (l>>4)*8.
#pragma unroll
  for (int nt = 0; nt < 4; ++nt) {
    const int colg = colBase + nt * 16 + ml;
    const float bv = bias[colg];
#pragma unroll
    for (int mt = 0; mt < 2; ++mt) {
#pragma unroll
      for (int v = 0; v < 8; ++v) {
        const int row = rowBase + mt * 16 + half * 8 + v;
        if (row < M)
          C[(size_t)row * ldc + coff + colg] = alpha * (acc[mt][nt][v] + bv);
      }
    }
  }
}

// ---------------------------------------------------------------------------
// SpMM scatter: Y[row[e], yoff + f] += scale * val[e] * X[col[e], f], f in 0..127
// One wave per edge; float4 gather + 4 global f32 atomics per lane.
// ---------------------------------------------------------------------------
__global__ void __launch_bounds__(256) spmm_atomic(
    const int* __restrict__ rows, const int* __restrict__ cols,
    const float* __restrict__ vals, const float* __restrict__ X,
    float* __restrict__ Y, int ldy, int yoff, float scale, int E)
{
  const int lane = threadIdx.x & 31;
  const int e = (int)((blockIdx.x * blockDim.x + threadIdx.x) >> 5);
  if (e >= E) return;
  const int r = rows[e];
  const int c = cols[e];
  const float v = vals[e] * scale;
  const float4 xv = reinterpret_cast<const float4*>(X + (size_t)c * 128)[lane];
  float* yp = Y + (size_t)r * ldy + yoff + lane * 4;
  atomicAdd(yp + 0, v * xv.x);
  atomicAdd(yp + 1, v * xv.y);
  atomicAdd(yp + 2, v * xv.z);
  atomicAdd(yp + 3, v * xv.w);
}

// ---------------------------------------------------------------------------
// BatchNorm (batch statistics over N rows, 384 columns)
// ---------------------------------------------------------------------------
__global__ void __launch_bounds__(256) bn_partial(
    const float* __restrict__ H, int N, float* __restrict__ sums,
    float* __restrict__ sumsq, int rowsPerBlock)
{
  __shared__ float ls[384];
  __shared__ float lq[384];
  const int tid = threadIdx.x;
  for (int c = tid; c < 384; c += 256) { ls[c] = 0.f; lq[c] = 0.f; }
  __syncthreads();
  const int r0 = blockIdx.x * rowsPerBlock;
  int r1 = r0 + rowsPerBlock; if (r1 > N) r1 = N;
  if (r1 > r0) {
    const long long base = (long long)r0 * 384;
    const long long total = (long long)(r1 - r0) * 384;
    for (long long i = tid; i < total; i += 256) {
      const float h = H[base + i];
      const int c = (int)(i % 384);
      atomicAdd(&ls[c], h);         // ds_add_f32
      atomicAdd(&lq[c], h * h);
    }
  }
  __syncthreads();
  for (int c = tid; c < 384; c += 256) {
    atomicAdd(&sums[c], ls[c]);
    atomicAdd(&sumsq[c], lq[c]);
  }
}

__global__ void __launch_bounds__(256) bn_finalize(
    const float* __restrict__ sums, const float* __restrict__ sumsq,
    const float* __restrict__ gamma, const float* __restrict__ beta,
    float* __restrict__ scl, float* __restrict__ shf, int ncol, float invN)
{
  const int c = blockIdx.x * blockDim.x + threadIdx.x;
  if (c < ncol) {
    const float mu  = sums[c] * invN;
    const float var = sumsq[c] * invN - mu * mu;
    const float rs  = rsqrtf(var + 1e-5f);
    const float s   = gamma[c] * rs;
    scl[c] = s;
    shf[c] = beta[c] - mu * s;
  }
}

__global__ void __launch_bounds__(256) bn_apply_relu_bf16(
    const float* __restrict__ H, const float* __restrict__ scl,
    const float* __restrict__ shf, unsigned short* __restrict__ out,
    long long total)
{
  long long i = (long long)blockIdx.x * blockDim.x + threadIdx.x;
  long long s = (long long)gridDim.x * blockDim.x;
  for (; i < total; i += s) {
    const int c = (int)(i % 384);
    float y = scl[c] * H[i] + shf[c];
    y = fmaxf(y, 0.f);
    out[i] = f2bf(y);
  }
}

// ---------------------------------------------------------------------------
// Host-side launcher
// ---------------------------------------------------------------------------
extern "C" void kernel_launch(void* const* d_in, const int* in_sizes, int n_in,
                              void* d_out, int out_size, void* d_ws, size_t ws_size,
                              hipStream_t stream)
{
  const float* x      = (const float*)d_in[0];
  const int*   row    = (const int*)d_in[1];
  const int*   col    = (const int*)d_in[2];
  const float* val    = (const float*)d_in[3];
  const int*   nd_row = (const int*)d_in[4];
  const int*   nd_col = (const int*)d_in[5];
  const float* nd_val = (const float*)d_in[6];
  const float* l1W[3] = {(const float*)d_in[7],  (const float*)d_in[9],  (const float*)d_in[11]};
  const float* l1b[3] = {(const float*)d_in[8],  (const float*)d_in[10], (const float*)d_in[12]};
  const float* l2W[3] = {(const float*)d_in[13], (const float*)d_in[15], (const float*)d_in[17]};
  const float* l2b[3] = {(const float*)d_in[14], (const float*)d_in[16], (const float*)d_in[18]};
  const float* gamma  = (const float*)d_in[19];
  const float* beta   = (const float*)d_in[20];
  const float* fpW    = (const float*)d_in[21];
  const float* fpb    = (const float*)d_in[22];
  float* out = (float*)d_out;

  const int N = in_sizes[0] / 128;   // 50000 nodes
  const int E = in_sizes[1];         // 800000 edges

  // ---- carve workspace ----
  size_t off = 0;
  auto carve = [&](size_t bytes) -> void* {
    void* p = (char*)d_ws + off;
    off = (off + bytes + 255) & ~(size_t)255;
    return p;
  };
  unsigned short* xb    = (unsigned short*)carve((size_t)N * 128 * 2);       // x in bf16
  unsigned short* w1b   = (unsigned short*)carve((size_t)3 * 128 * 128 * 2); // l1 weights bf16
  unsigned short* w2b   = (unsigned short*)carve((size_t)3 * 128 * 384 * 2); // l2 weights bf16
  unsigned short* fpwb  = (unsigned short*)carve((size_t)128 * 384 * 2);     // fp weights bf16
  float*          T0    = (float*)carve((size_t)N * 128 * 4);                // hop projection
  float*          T1    = (float*)carve((size_t)N * 128 * 4);                // 2-hop intermediate
  float*          H     = (float*)carve((size_t)N * 384 * 4);                // concat output (f32)
  unsigned short* Hbf   = (unsigned short*)carve((size_t)N * 384 * 2);       // concat (bf16)
  float*          sums  = (float*)carve(384 * 4);
  float*          sumsq = (float*)carve(384 * 4);
  float*          scl   = (float*)carve(384 * 4);
  float*          shf   = (float*)carve(384 * 4);
  (void)ws_size; (void)n_in; (void)out_size;

  // ---- launch helpers ----
  auto fill = [&](float* p, long long n) {
    fill_f32<<<dim3(2048), dim3(256), 0, stream>>>(p, n, 0.f);
  };
  auto cvt = [&](const float* in, unsigned short* o, long long n) {
    int blocks = (int)((n + 1023) / 1024); if (blocks > 8192) blocks = 8192; if (blocks < 1) blocks = 1;
    cvt_f32_bf16<<<dim3(blocks), dim3(256), 0, stream>>>(in, o, n);
  };
  auto gemm = [&](const unsigned short* A, const unsigned short* W, const float* b,
                  float* C, int M, int K, int ldc, int coff, float alpha) {
    gemm_bf16_n128<<<dim3((M + 127) / 128), dim3(256), 0, stream>>>(
        (const bf16_t*)A, (const bf16_t*)W, b, C, M, K, ldc, coff, alpha);
  };
  auto spmm = [&](const int* r, const int* c, const float* v, const float* X,
                  float* Y, int ldy, int yoff, float scale) {
    spmm_atomic<<<dim3((E + 7) / 8), dim3(256), 0, stream>>>(r, c, v, X, Y, ldy, yoff, scale, E);
  };

  // ---- precision conversion of operands ----
  cvt(x, xb, (long long)N * 128);
  for (int h = 0; h < 3; ++h) cvt(l1W[h], w1b + (size_t)h * 128 * 128, 128 * 128);
  for (int h = 0; h < 3; ++h) cvt(l2W[h], w2b + (size_t)h * 128 * 384, 128 * 384);
  cvt(fpW, fpwb, 128 * 384);

  // ======================= Layer 1 (MixHop + BN + ReLU) =====================
  fill(H, (long long)N * 384);
  // hop0: 1.5 * (x @ W0^T + b0)  (shared between both adjacency terms)
  gemm(xb, w1b + 0 * 128 * 128, l1b[0], H, N, 128, 384, 0, 1.5f);
  // hop1: A*P1 + 0.5*ND*P1
  gemm(xb, w1b + 1 * 128 * 128, l1b[1], T0, N, 128, 128, 0, 1.0f);
  spmm(row, col, val, T0, H, 384, 128, 1.0f);
  spmm(nd_row, nd_col, nd_val, T0, H, 384, 128, 0.5f);
  // hop2: A*A*P2 + 0.5*ND*ND*P2
  gemm(xb, w1b + 2 * 128 * 128, l1b[2], T0, N, 128, 128, 0, 1.0f);
  fill(T1, (long long)N * 128);
  spmm(row, col, val, T0, T1, 128, 0, 1.0f);
  spmm(row, col, val, T1, H, 384, 256, 1.0f);
  fill(T1, (long long)N * 128);
  spmm(nd_row, nd_col, nd_val, T0, T1, 128, 0, 1.0f);
  spmm(nd_row, nd_col, nd_val, T1, H, 384, 256, 0.5f);
  // BatchNorm (batch stats) + ReLU + bf16 convert
  fill(sums, 384); fill(sumsq, 384);
  bn_partial<<<dim3((N + 255) / 256), dim3(256), 0, stream>>>(H, N, sums, sumsq, 256);
  bn_finalize<<<dim3(2), dim3(256), 0, stream>>>(sums, sumsq, gamma, beta, scl, shf, 384, 1.0f / (float)N);
  bn_apply_relu_bf16<<<dim3(8192), dim3(256), 0, stream>>>(H, scl, shf, Hbf, (long long)N * 384);

  // ======================= Layer 2 (MixHop, no BN) ==========================
  fill(H, (long long)N * 384);
  gemm(Hbf, w2b + 0 * (size_t)128 * 384, l2b[0], H, N, 384, 384, 0, 1.5f);
  gemm(Hbf, w2b + 1 * (size_t)128 * 384, l2b[1], T0, N, 384, 128, 0, 1.0f);
  spmm(row, col, val, T0, H, 384, 128, 1.0f);
  spmm(nd_row, nd_col, nd_val, T0, H, 384, 128, 0.5f);
  gemm(Hbf, w2b + 2 * (size_t)128 * 384, l2b[2], T0, N, 384, 128, 0, 1.0f);
  fill(T1, (long long)N * 128);
  spmm(row, col, val, T0, T1, 128, 0, 1.0f);
  spmm(row, col, val, T1, H, 384, 256, 1.0f);
  fill(T1, (long long)N * 128);
  spmm(nd_row, nd_col, nd_val, T0, T1, 128, 0, 1.0f);
  spmm(nd_row, nd_col, nd_val, T1, H, 384, 256, 0.5f);

  // ======================= Final projection =================================
  cvt(H, Hbf, (long long)N * 384);
  gemm(Hbf, fpwb, fpb, out, N, 384, 128, 0, 1.0f);
}
